// Prealign_68616397521540
// MI455X (gfx1250) — compile-verified, matrix operand order checked by reading, not run on hardware
//
#include <hip/hip_runtime.h>
#include <hip/hip_bf16.h>

typedef __attribute__((ext_vector_type(16))) _Float16 v16h;
typedef __attribute__((ext_vector_type(8)))  float    v8f;

#define EMBED 768
#define HIDDEN 768
#define LD_DOC 1024
#define LQ 128
#define NB 32
#define WAVES 8
#define MT 32   // doc rows per workgroup in K2 (two 16-row WMMA subtiles)

// ---------------------------------------------------------------------------
// Fragment loaders for V_WMMA_F32_16X16X32_F16 (ISA 7.12.2 layouts).
// A matrix 16x32 f16: lanes 0-15 hold M=lane with K={k0..k0+7, k0+16..k0+23},
// lanes 16-31 hold M=lane-16 with K={k0+8..k0+15, k0+24..k0+31}.
// B matrix 32x16 mirrors this with lane&15 = column N, so a row-major [N][K]
// source (e.g. W[h][e], query_o[q][h], qeT[e][q]) loads with the same gather.
// ---------------------------------------------------------------------------
__device__ __forceinline__ v16h frag_f16(const _Float16* mat, int ld,
                                         int row_base, int k0) {
  const int lane = threadIdx.x & 31;
  const int r    = lane & 15;
  const int off  = (lane >> 4) * 8;
  const _Float16* p = mat + (size_t)(row_base + r) * ld + k0 + off;
  v16h f;
  ((uint4*)&f)[0] = *(const uint4*)(p);        // K = k0+off .. +7
  ((uint4*)&f)[1] = *(const uint4*)(p + 16);   // K = k0+16+off .. +7
  return f;
}

__device__ __forceinline__ v16h frag_f32(const float* mat, int ld,
                                         int row_base, int k0) {
  const int lane = threadIdx.x & 31;
  const int r    = lane & 15;
  const int off  = (lane >> 4) * 8;
  const float* p = mat + (size_t)(row_base + r) * ld + k0 + off;
  v16h f;
#pragma unroll
  for (int i = 0; i < 8; ++i) {
    f[i]     = (_Float16)p[i];
    f[i + 8] = (_Float16)p[i + 16];
  }
  return f;
}

__device__ __forceinline__ v8f wmma_f16(v16h a, v16h b, v8f c) {
  return __builtin_amdgcn_wmma_f32_16x16x32_f16(false, a, false, b,
                                                (short)0, c, false, false);
}

// ---------------------------------------------------------------------------
// K0: W f32 -> f16
// ---------------------------------------------------------------------------
__global__ __launch_bounds__(256) void k0_cvt_w(const float* __restrict__ W,
                                                _Float16* __restrict__ Wh,
                                                int n) {
  int i = blockIdx.x * 256 + threadIdx.x;
  if (i < n) Wh[i] = (_Float16)W[i];
}

// ---------------------------------------------------------------------------
// K1: query_o = relu(query_embed @ W^T + b) -> f16 [B][128][768]
//     qeT     = query_embed^T f16            -> [B][768][128]
// grid (LQ/16, NB), 256 threads (8 waves). Each wave owns 6 h-tiles.
// ---------------------------------------------------------------------------
__global__ __launch_bounds__(256) void k1_query(const float* __restrict__ qe,
                                                const _Float16* __restrict__ Wh,
                                                const float* __restrict__ bias,
                                                _Float16* __restrict__ q_o,
                                                _Float16* __restrict__ qeT) {
  const int bb    = blockIdx.y;
  const int qbase = blockIdx.x * 16;
  const int wave  = threadIdx.x >> 5;
  const int lane  = threadIdx.x & 31;
  const float* qe_b = qe + (size_t)bb * LQ * EMBED;

  v8f acc[6] = {};
#pragma unroll 4
  for (int k0 = 0; k0 < EMBED; k0 += 32) {
    v16h a = frag_f32(qe_b, EMBED, qbase, k0);
#pragma unroll
    for (int j = 0; j < 6; ++j) {
      const int hbase = (j * WAVES + wave) * 16;
      v16h bf = frag_f16(Wh, EMBED, hbase, k0);   // B[e][h] = W[h][e]
      acc[j] = wmma_f16(a, bf, acc[j]);
    }
  }

  // bias + relu + store f16
  const int n  = lane & 15;
  const int hi = lane >> 4;
  _Float16* qo_b = q_o + (size_t)bb * LQ * HIDDEN;
#pragma unroll
  for (int j = 0; j < 6; ++j) {
    const int hbase = (j * WAVES + wave) * 16;
    const float bn = bias[hbase + n];
#pragma unroll
    for (int i = 0; i < 8; ++i) {
      const int m = i + hi * 8;
      qo_b[(size_t)(qbase + m) * HIDDEN + hbase + n] =
          (_Float16)fmaxf(acc[j][i] + bn, 0.0f);
    }
  }

  // transpose the 16 query rows into qeT[b][e][q] (f16)
  _Float16* qeT_b = qeT + (size_t)bb * EMBED * LQ;
  for (int idx = threadIdx.x; idx < EMBED * 16; idx += 256) {
    const int e  = idx >> 4;
    const int qq = idx & 15;
    qeT_b[(size_t)e * LQ + qbase + qq] = (_Float16)qe_b[(size_t)(qbase + qq) * EMBED + e];
  }
}

// ---------------------------------------------------------------------------
// K2: fused per MT=32 doc rows (two 16-row WMMA subtiles, so every W/V
// B-fragment feeds two WMMAs -> halves L2 traffic for W):
//   doc_o(LDS f16) -> scores(LDS f32) -> softmax -> attn(LDS f16) -> out f32
// grid (LD/MT, NB), 256 threads (8 waves).
// ---------------------------------------------------------------------------
__global__ __launch_bounds__(256) void k2_doc(const float* __restrict__ doc,
                                              const _Float16* __restrict__ Wh,
                                              const float* __restrict__ bias,
                                              const _Float16* __restrict__ q_o,
                                              const _Float16* __restrict__ qeT,
                                              float* __restrict__ out) {
  __shared__ _Float16 s_doco[MT * HIDDEN];   // 48 KB
  __shared__ float    s_scores[MT * LQ];     // 16 KB
  __shared__ _Float16 s_attn[MT * LQ];       //  8 KB

  const int bb    = blockIdx.y;
  const int dbase = blockIdx.x * MT;
  const int wave  = threadIdx.x >> 5;
  const int lane  = threadIdx.x & 31;
  const int n     = lane & 15;
  const int hi    = lane >> 4;

  const float*    doc_b = doc + (size_t)bb * LD_DOC * EMBED;
  const _Float16* qo_b  = q_o + (size_t)bb * LQ * HIDDEN;
  const _Float16* qeT_b = qeT + (size_t)bb * EMBED * LQ;

  // ---- stage A: doc_o[32,768] = relu(doc[32,:] @ W^T + b); 48 h-tiles/8 waves
  {
    v8f acc0[6] = {};
    v8f acc1[6] = {};
#pragma unroll 2
    for (int k0 = 0; k0 < EMBED; k0 += 32) {
      v16h a0 = frag_f32(doc_b, EMBED, dbase, k0);
      v16h a1 = frag_f32(doc_b, EMBED, dbase + 16, k0);
#pragma unroll
      for (int j = 0; j < 6; ++j) {
        const int hbase = (j * WAVES + wave) * 16;
        v16h bf = frag_f16(Wh, EMBED, hbase, k0);
        acc0[j] = wmma_f16(a0, bf, acc0[j]);
        acc1[j] = wmma_f16(a1, bf, acc1[j]);
      }
    }
#pragma unroll
    for (int j = 0; j < 6; ++j) {
      const int hbase = (j * WAVES + wave) * 16;
      const float bn = bias[hbase + n];
#pragma unroll
      for (int i = 0; i < 8; ++i) {
        const int m = i + hi * 8;
        s_doco[m * HIDDEN + hbase + n] =
            (_Float16)fmaxf(acc0[j][i] + bn, 0.0f);
        s_doco[(m + 16) * HIDDEN + hbase + n] =
            (_Float16)fmaxf(acc1[j][i] + bn, 0.0f);
      }
    }
  }
  __syncthreads();

  // ---- stage B: scores[32, wave*16..+15] = doc_o @ query_o^T
  {
    const int qbase = wave * 16;
    v8f sc0 = {};
    v8f sc1 = {};
#pragma unroll 4
    for (int k0 = 0; k0 < HIDDEN; k0 += 32) {
      v16h a0 = frag_f16(s_doco, HIDDEN, 0, k0);       // LDS A operand, rows 0-15
      v16h a1 = frag_f16(s_doco, HIDDEN, 16, k0);      // rows 16-31
      v16h bq = frag_f16(qo_b, HIDDEN, qbase, k0);     // B[h][q] = query_o[q][h]
      sc0 = wmma_f16(a0, bq, sc0);
      sc1 = wmma_f16(a1, bq, sc1);
    }
#pragma unroll
    for (int i = 0; i < 8; ++i) {
      const int m = i + hi * 8;
      s_scores[m * LQ + qbase + n]        = sc0[i];
      s_scores[(m + 16) * LQ + qbase + n] = sc1[i];
    }
  }
  __syncthreads();

  // ---- stage C: softmax over q (128) per doc row; attn stored f16
  if (threadIdx.x < MT) {
    const int m = threadIdx.x;
    float mx = -3.0e38f;
    for (int q = 0; q < LQ; ++q) mx = fmaxf(mx, s_scores[m * LQ + q]);
    float sum = 0.0f;
    for (int q = 0; q < LQ; ++q) {
      const float e = __expf(s_scores[m * LQ + q] - mx);
      s_scores[m * LQ + q] = e;
      sum += e;
    }
    const float inv = 1.0f / sum;
    for (int q = 0; q < LQ; ++q)
      s_attn[m * LQ + q] = (_Float16)(s_scores[m * LQ + q] * inv);
  }
  __syncthreads();

  // ---- stage D: out[32, :] = attn @ query_embed, 48 e-tiles / 8 waves
  float* out_b = out + ((size_t)bb * LD_DOC + dbase) * EMBED;
#pragma unroll
  for (int j = 0; j < 6; ++j) {
    const int ebase = (j * WAVES + wave) * 16;
    v8f oc0 = {};
    v8f oc1 = {};
#pragma unroll
    for (int k0 = 0; k0 < LQ; k0 += 32) {
      v16h a0 = frag_f16(s_attn, LQ, 0, k0);
      v16h a1 = frag_f16(s_attn, LQ, 16, k0);
      v16h bv = frag_f16(qeT_b, LQ, ebase, k0);        // B[q][e] = qeT[e][q]
      oc0 = wmma_f16(a0, bv, oc0);
      oc1 = wmma_f16(a1, bv, oc1);
    }
#pragma unroll
    for (int i = 0; i < 8; ++i) {
      const int m = i + hi * 8;
      out_b[(size_t)m * EMBED + ebase + n]        = oc0[i];
      out_b[(size_t)(m + 16) * EMBED + ebase + n] = oc1[i];
    }
  }
}

// ---------------------------------------------------------------------------
extern "C" void kernel_launch(void* const* d_in, const int* in_sizes, int n_in,
                              void* d_out, int out_size, void* d_ws, size_t ws_size,
                              hipStream_t stream) {
  (void)in_sizes; (void)n_in; (void)out_size; (void)ws_size;
  const float* doc = (const float*)d_in[0];   // [32,1024,768]
  const float* qe  = (const float*)d_in[1];   // [32,128,768]
  const float* W   = (const float*)d_in[2];   // [768,768]
  const float* b   = (const float*)d_in[3];   // [768]
  float* out = (float*)d_out;                 // [32,1024,768]

  // workspace layout (f16): W_h | query_o | query_embed^T
  char* ws = (char*)d_ws;
  _Float16* Wh  = (_Float16*)ws;                                  // 768*768
  _Float16* q_o = (_Float16*)(ws + (size_t)HIDDEN * EMBED * 2);   // 32*128*768
  _Float16* qeT = (_Float16*)(ws + (size_t)HIDDEN * EMBED * 2
                                 + (size_t)NB * LQ * HIDDEN * 2); // 32*768*128

  const int nW = HIDDEN * EMBED;
  k0_cvt_w<<<(nW + 255) / 256, 256, 0, stream>>>(W, Wh, nW);
  k1_query<<<dim3(LQ / 16, NB), 256, 0, stream>>>(qe, Wh, b, q_o, qeT);
  k2_doc<<<dim3(LD_DOC / MT, NB), 256, 0, stream>>>(doc, Wh, b, q_o, qeT, out);
}
